// RNNLayer_54331336294992
// MI455X (gfx1250) — matrix operand order repaired and checked by reading, not compiled, not run on hardware
//
#include <hip/hip_runtime.h>
#include <math.h>

typedef __attribute__((ext_vector_type(16))) __bf16 v16bf;
typedef __attribute__((ext_vector_type(8)))  __bf16 v8bf;
typedef __attribute__((ext_vector_type(8)))  float  v8f;
typedef __attribute__((ext_vector_type(4)))  float  v4f;

#define B_  64
#define T_  1024
#define IN_ 512
#define H_  512

// ---------------------------------------------------------------------------
// fp32 -> bf16 conversion (weights, h0); also zeroes the 4 barrier counters.
// ---------------------------------------------------------------------------
__global__ __launch_bounds__(256)
void cvt_f32_bf16(const float* __restrict__ in, __bf16* __restrict__ out,
                  int n, unsigned* __restrict__ bar)
{
    int i = blockIdx.x * 256 + threadIdx.x;
    if (i < n) out[i] = (__bf16)in[i];
    if (bar && blockIdx.x == 0 && threadIdx.x < 4) bar[threadIdx.x] = 0u;
}

// ---------------------------------------------------------------------------
// Phase 1: Wx = X @ W^T + b   (M = B*T = 65536, N = H = 512, K = IN = 512)
// grid = (8 n-blocks, 512 m-blocks), 256 threads = 8 waves.
// Each wave computes a 16x64 tile (4 N-tiles) of the output; WG tile 128x64.
// bf16 WMMA, fp32 accumulate, result (pre-activation) written to d_out.
// ---------------------------------------------------------------------------
__global__ __launch_bounds__(256)
void wx_gemm(const float* __restrict__ X,     // [B*T, IN]
             const __bf16* __restrict__ Wbf,  // [H, IN] bf16
             const float* __restrict__ Wb,    // [H]
             float* __restrict__ Out)         // [B*T, H]
{
    const int wave = threadIdx.x >> 5;
    const int lane = threadIdx.x & 31;
    const int l16  = lane & 15;
    const int hi   = lane >> 4;      // half-wave select
    const int hi8  = hi * 8;

    const int m0 = blockIdx.y * 128 + wave * 16;   // row block of this wave
    const int n0 = blockIdx.x * 64;                // col block of this WG
    const int arow = m0 + l16;                     // A-matrix row for this lane

    v8f acc[4] = {};

    #pragma unroll 2
    for (int k = 0; k < IN_; k += 32) {
        // ---- A fragment: 16x32 bf16, converted from fp32 X on the fly ----
        const float* ap = X + (size_t)arow * IN_ + k + hi * 16;
        v4f a0 = *(const v4f*)(ap);
        v4f a1 = *(const v4f*)(ap + 4);
        v4f a2 = *(const v4f*)(ap + 8);
        v4f a3 = *(const v4f*)(ap + 12);
        v16bf A;
        #pragma unroll
        for (int e = 0; e < 4; ++e) {
            A[e]      = (__bf16)a0[e];
            A[4 + e]  = (__bf16)a1[e];
            A[8 + e]  = (__bf16)a2[e];
            A[12 + e] = (__bf16)a3[e];
        }
        // ---- 4 B fragments (32x16 bf16 each) straight from L2 ----
        #pragma unroll
        for (int t = 0; t < 4; ++t) {
            const __bf16* bp = Wbf + (size_t)(n0 + t * 16 + l16) * IN_ + k + hi * 16;
            v16bf Bf = *(const v16bf*)bp;
            acc[t] = __builtin_amdgcn_wmma_f32_16x16x32_bf16(
                         false, A, false, Bf, (short)0, acc[t], false, false);
        }
    }

    // ---- bias + store (pre-activation) ----
    #pragma unroll
    for (int t = 0; t < 4; ++t) {
        const int col  = n0 + t * 16 + l16;
        const float bv = Wb[col];
        #pragma unroll
        for (int r = 0; r < 8; ++r) {
            Out[(size_t)(m0 + r + hi8) * H_ + col] = acc[t][r] + bv;
        }
    }
}

// ---------------------------------------------------------------------------
// Phase 2: persistent sequential scan over T.
// grid = 16 WGs: blockIdx = bblk*4 + nblk. Each WG: 16 batch rows x 128 cols.
// 8 waves/WG, each wave owns a 16-col slice of U, kept in 128 VGPRs (bf16)
// for all 1024 steps. h ping-pongs as bf16 through a small global buffer.
// The 4 WGs of a batch block sync per step with an L2 atomic counter.
// Wx is read from d_out and overwritten in place with h_t.
// ---------------------------------------------------------------------------
__global__ __launch_bounds__(256)
void rnn_scan(float* __restrict__ Y,            // [B,T,H]: Wx in, h out
              const __bf16* __restrict__ Ubf,   // [H,H] bf16 (n,k)
              const float* __restrict__ Ub,     // [H]
              __bf16* __restrict__ hbuf,        // [2][B][H] bf16 ping-pong
              unsigned* __restrict__ bar)       // [4] per-batch-block counter
{
    const int wave = threadIdx.x >> 5;
    const int lane = threadIdx.x & 31;
    const int l16  = lane & 15;
    const int hi   = lane >> 4;
    const int hi8  = hi * 8;

    const int bblk = blockIdx.x >> 2;            // 0..3 : batch block (16 rows)
    const int nblk = blockIdx.x & 3;             // 0..3 : 128-col block
    const int n0   = nblk * 128 + wave * 16;     // wave's N-tile
    const int col  = n0 + l16;                   // this lane's output column
    const int brow = bblk * 16;
    const float ubias = Ub[col];

    // U tile resident in registers: 16 k-fragments (K=512), 8 VGPRs each.
    v16bf Bf[16];
    #pragma unroll
    for (int kf = 0; kf < 16; ++kf)
        Bf[kf] = *(const v16bf*)(Ubf + (size_t)col * H_ + kf * 32 + hi * 16);

    const size_t rstride = (size_t)T_ * H_;

    #pragma unroll 1
    for (int t = 0; t < T_; ++t) {
        const __bf16* hprev = hbuf + (size_t)(t & 1) * (B_ * H_);
        __bf16*       hnext = hbuf + (size_t)((t + 1) & 1) * (B_ * H_);

        // acc = Wx[:, t, col] + U_b[col]
        v8f acc;
        #pragma unroll
        for (int r = 0; r < 8; ++r)
            acc[r] = Y[(size_t)(brow + r + hi8) * rstride + (size_t)t * H_ + col] + ubias;

        // acc += h_prev @ U^T  (K = 512, 16 WMMAs, U from registers)
        #pragma unroll
        for (int kf = 0; kf < 16; ++kf) {
            const __bf16* hp = hprev + (size_t)(brow + l16) * H_ + kf * 32 + hi * 16;
            v8bf a0 = *(const v8bf*)hp;
            v8bf a1 = *(const v8bf*)(hp + 8);
            v16bf A = __builtin_shufflevector(a0, a1,
                        0,1,2,3,4,5,6,7,8,9,10,11,12,13,14,15);
            acc = __builtin_amdgcn_wmma_f32_16x16x32_bf16(
                      false, A, false, Bf[kf], (short)0, acc, false, false);
        }

        // tanh, store fp32 to output (in place over Wx) + bf16 for next step
        #pragma unroll
        for (int r = 0; r < 8; ++r) {
            float v = tanhf(acc[r]);
            Y[(size_t)(brow + r + hi8) * rstride + (size_t)t * H_ + col] = v;
            hnext[(size_t)(brow + r + hi8) * H_ + col] = (__bf16)v;
        }

        // release stores, then arrive + spin on the batch-block counter
        __threadfence();
        __syncthreads();
        if (threadIdx.x == 0) {
            atomicAdd(bar + bblk, 1u);
            const unsigned tgt = 4u * (unsigned)(t + 1);
            while (atomicAdd(bar + bblk, 0u) < tgt)
                __builtin_amdgcn_s_sleep(1);
        }
        __syncthreads();
        __threadfence();   // acquire: invalidate before reading peers' h
    }
}

// ---------------------------------------------------------------------------
extern "C" void kernel_launch(void* const* d_in, const int* in_sizes, int n_in,
                              void* d_out, int out_size, void* d_ws, size_t ws_size,
                              hipStream_t stream)
{
    const float* X  = (const float*)d_in[0];   // [B,T,IN]
    const float* h0 = (const float*)d_in[1];   // [B,H]
    const float* Ww = (const float*)d_in[2];   // [H,IN]
    const float* Wb = (const float*)d_in[3];   // [H]
    const float* Uw = (const float*)d_in[4];   // [H,H]
    const float* Ub = (const float*)d_in[5];   // [H]
    float* Y = (float*)d_out;                  // [B,T,H]

    char* ws = (char*)d_ws;
    __bf16*   Wbf  = (__bf16*)ws;                          // 512 KB
    __bf16*   Ubf  = Wbf + (size_t)H_ * IN_;               // 512 KB
    __bf16*   hbuf = Ubf + (size_t)H_ * H_;                // 128 KB ping-pong
    unsigned* bar  = (unsigned*)(hbuf + 2 * (size_t)B_ * H_);

    // Weight / h0 conversion to bf16; third launch also zeroes barriers.
    cvt_f32_bf16<<<(H_ * IN_ + 255) / 256, 256, 0, stream>>>(Ww, Wbf, H_ * IN_, nullptr);
    cvt_f32_bf16<<<(H_ * H_  + 255) / 256, 256, 0, stream>>>(Uw, Ubf, H_ * H_,  nullptr);
    cvt_f32_bf16<<<(B_ * H_  + 255) / 256, 256, 0, stream>>>(h0, hbuf, B_ * H_, bar);

    // Phase 1: input projection into d_out (pre-activations).
    dim3 g1(H_ / 64, (B_ * T_) / 128);
    wx_gemm<<<g1, 256, 0, stream>>>(X, Wbf, Wb, Y);

    // Phase 2: persistent recurrent scan (16 WGs, grid-synced per step).
    rnn_scan<<<16, 256, 0, stream>>>(Y, Ubf, Ub, hbuf, bar);
}
